// BinaryConv_51719996178703
// MI455X (gfx1250) — matrix operand order repaired and checked by reading, not compiled, not run on hardware
//
#include <hip/hip_runtime.h>
#include <hip/hip_bf16.h>

typedef __attribute__((ext_vector_type(16))) _Float16 v16h;
typedef __attribute__((ext_vector_type(8)))  _Float16 v8h;
typedef __attribute__((ext_vector_type(8)))  float    v8f;

// Problem dims (fixed by reference)
#define NB   16          // batch
#define CI   256         // in channels
#define CO   256         // out channels
#define HW   56          // H = W
#define HP   58          // padded H/W
#define KTAPS 9          // 3x3
#define KTOT (KTAPS * CI)        // 2304
#define NPIX (NB * HW * HW)      // 50176
#define WELEM (CO * CI * KTAPS)  // 589824
#define NCHUNK 72                // 9 taps * 8 ci-chunks of 32
#define PIXT 64                  // pixels per block tile

// -------- workspace layout --------
// [0,256)                : float sum(|w|)
// [256, 256+WELEM*2)     : bw f16, [CO][(kh*3+kw)*256 + ci]
// [.., + NB*HP*HP*CI*2)  : x padded NHWC f16
#define WS_BW_OFF   256
#define WS_XH_OFF   (256 + WELEM * 2)   // 1,179,904 (256-aligned)

// ---------------- kernel 0: zero the reduction scalar ----------------
__global__ void k_zero(float* s) { s[0] = 0.0f; }

// ---------------- kernel 1: sum |w| ----------------
__global__ void k_abs_sum(const float* __restrict__ w, float* __restrict__ s) {
    __shared__ float red[256];
    float acc = 0.0f;
    for (int i = blockIdx.x * blockDim.x + threadIdx.x; i < WELEM;
         i += gridDim.x * blockDim.x)
        acc += fabsf(w[i]);
    red[threadIdx.x] = acc;
    __syncthreads();
    for (int off = 128; off > 0; off >>= 1) {
        if (threadIdx.x < off) red[threadIdx.x] += red[threadIdx.x + off];
        __syncthreads();
    }
    if (threadIdx.x == 0) atomicAdd(s, red[0]);
}

// ---------------- kernel 2: binarize + repack weights ----------------
__global__ void k_binarize(const float* __restrict__ w, _Float16* __restrict__ bw) {
    int i = blockIdx.x * blockDim.x + threadIdx.x;
    if (i >= WELEM) return;
    int m  = i / KTOT;
    int k  = i - m * KTOT;
    int r  = k >> 8;          // tap index 0..8
    int ci = k & 255;
    float v = w[(m * CI + ci) * KTAPS + r];
    float s = (v > 0.0f) ? 1.0f : ((v < 0.0f) ? -1.0f : 0.0f);
    bw[i] = (_Float16)s;
}

// ---------------- kernel 3: x NCHW f32 -> padded NHWC f16 ----------------
// ci fastest in tid -> 64B-coalesced f16 writes; strided reads are L2-backed
// (entire 51 MB x fits in the 192 MB L2).
__global__ void k_pack_x(const float* __restrict__ x, _Float16* __restrict__ xh) {
    long long i = (long long)blockIdx.x * blockDim.x + threadIdx.x;
    const long long total = (long long)NB * HP * HP * CI;
    if (i >= total) return;
    int ci = (int)(i & 255);
    long long t = i >> 8;
    int wp = (int)(t % HP); t /= HP;
    int hp = (int)(t % HP);
    int n  = (int)(t / HP);
    float v = 0.0f;
    if (hp >= 1 && hp <= HW && wp >= 1 && wp <= HW)
        v = x[(((long long)n * CI + ci) * HW + (hp - 1)) * HW + (wp - 1)];
    xh[i] = (_Float16)v;
}

// ------- async global -> LDS (CDNA5, ASYNCcnt-tracked, 16B per lane) -------
__device__ __forceinline__ void async_ld_b128(unsigned lds_byte, const void* gptr) {
    asm volatile("global_load_async_to_lds_b128 %0, %1, off"
                 :: "v"(lds_byte), "v"(gptr) : "memory");
}
__device__ __forceinline__ void wait_async0() {
    asm volatile("s_wait_asynccnt 0x0" ::: "memory");
}

// ---------------- kernel 4: implicit-GEMM binary conv via WMMA ----------------
// block = 256 threads = 8 waves, each wave: 16(M) x 64(N) tile (4 accumulators
// sharing one A fragment). Block tile: 128 out-channels x 64 pixels. The
// 64pix x 32ci B chunk (4 KB) is DMA'd to LDS once per K-chunk (double-
// buffered, all 256 threads issue one b128) and shared by all 8 waves, so each
// asynccnt-wait + barrier covers 4 WMMAs per wave.
__global__ __launch_bounds__(256)
void k_conv_wmma(const _Float16* __restrict__ bw,
                 const _Float16* __restrict__ xh,
                 const float* __restrict__ sumabs,
                 float* __restrict__ out) {
    __shared__ _Float16 smem[2][PIXT * 32];   // [buf][pixel 0..63][ci 0..31]

    const int tid  = threadIdx.x;
    const int lane = tid & 31;
    const int wave = tid >> 5;
    const int half = lane >> 4;   // 0/1
    const int l16  = lane & 15;

    const int m_base = blockIdx.y * 128 + wave * 16;    // out-channel tile base

    // this lane's four output pixels (N columns of the four B fragments)
    int pn[4], ph[4], pw[4];
    #pragma unroll
    for (int t = 0; t < 4; ++t) {
        const int pix = blockIdx.x * PIXT + t * 16 + l16;
        const int n = pix / (HW * HW), r = pix - n * (HW * HW);
        pn[t] = n; ph[t] = r / HW; pw[t] = r - (r / HW) * HW;
    }

    // A fragment rows: lane holds out-channel m_base + l16
    const _Float16* __restrict__ arow = bw + (long long)(m_base + l16) * KTOT;

    // DMA-issuing setup: 256 lanes x 16B = the 4 KB B chunk
    const int p = tid >> 2, q = tid & 3;     // pixel 0..63, 16B quarter 0..3
    const int pp = blockIdx.x * PIXT + p;
    const int nn = pp / (HW * HW), rr = pp - nn * (HW * HW);
    const int hh = rr / HW,        ww = rr - hh * HW;
    const _Float16* grow = xh + ((long long)(nn * HP + hh) * HP + ww) * CI + q * 8;
    const unsigned lds_dst =
        (unsigned)(size_t)(&smem[0][0]) + (unsigned)(p * 64 + q * 16);

    #define ISSUE_CHUNK(KK, BUF)                                               \
        do {                                                                   \
            const int r_ = (KK) >> 3, c8_ = (KK) & 7;                          \
            const int kh_ = r_ / 3, kw_ = r_ - kh_ * 3;                        \
            async_ld_b128(lds_dst + (unsigned)((BUF) * (PIXT * 32 * 2)),       \
                          (const void*)(grow + (kh_ * HP + kw_) * CI + c8_ * 32)); \
        } while (0)

    ISSUE_CHUNK(0, 0);

    v8f acc[4] = {{}, {}, {}, {}};

    #pragma unroll 4
    for (int kk = 0; kk < NCHUNK; ++kk) {
        const int buf = kk & 1;
        wait_async0();           // DMA into smem[buf] complete
        __syncthreads();         // ..visible to all waves; prev reads done
        if (kk + 1 < NCHUNK) ISSUE_CHUNK(kk + 1, buf ^ 1);

        const int r = kk >> 3, c8 = kk & 7;
        const _Float16* __restrict__ pA = arow + r * CI + c8 * 32;
        // prefetch next A fragment toward cache while this chunk computes
        if (kk + 1 < NCHUNK) {
            const int r2 = (kk + 1) >> 3, c82 = (kk + 1) & 7;
            __builtin_prefetch(arow + r2 * CI + c82 * 32, 0, 0);
        }
        // A fragment: 16 contiguous f16 (K = +half*16)
        v16h a = *(const v16h*)(pA + half * 16);

        // B fragments from LDS: 16 consecutive ci at each of this lane's pixels
        const _Float16* sb = &smem[buf][0];
        #pragma unroll
        for (int t = 0; t < 4; ++t) {
            v16h b = *(const v16h*)(sb + (t * 16 + l16) * 32 + half * 16);
            acc[t] = __builtin_amdgcn_wmma_f32_16x16x32_f16(
                false, a, false, b, (short)0, acc[t], false, false);
        }
    }

    // epilogue: scale by mean(|w|), store NCHW f32
    const float scale = sumabs[0] * (1.0f / (float)WELEM);
    #pragma unroll
    for (int rrow = 0; rrow < 8; ++rrow) {
        const int co = m_base + half * 8 + rrow;
        #pragma unroll
        for (int t = 0; t < 4; ++t) {
            out[(((long long)pn[t] * CO + co) * HW + ph[t]) * HW + pw[t]] =
                acc[t][rrow] * scale;
        }
    }
    #undef ISSUE_CHUNK
}

extern "C" void kernel_launch(void* const* d_in, const int* in_sizes, int n_in,
                              void* d_out, int out_size, void* d_ws, size_t ws_size,
                              hipStream_t stream) {
    const float* d_x = (const float*)d_in[0];
    const float* d_w = (const float*)d_in[1];
    float* d_o = (float*)d_out;

    char* ws = (char*)d_ws;
    float*     d_sum = (float*)ws;
    _Float16*  d_bw  = (_Float16*)(ws + WS_BW_OFF);
    _Float16*  d_xh  = (_Float16*)(ws + WS_XH_OFF);

    k_zero<<<1, 1, 0, stream>>>(d_sum);
    k_abs_sum<<<512, 256, 0, stream>>>(d_w, d_sum);
    k_binarize<<<(WELEM + 255) / 256, 256, 0, stream>>>(d_w, d_bw);
    {
        long long total = (long long)NB * HP * HP * CI;
        k_pack_x<<<(int)((total + 255) / 256), 256, 0, stream>>>(d_x, d_xh);
    }
    dim3 grid(NPIX / PIXT, CO / 128, 1);
    k_conv_wmma<<<grid, 256, 0, stream>>>(d_bw, d_xh, d_sum, d_o);
}
